// GNNLayerWithResidual_40802189312039
// MI455X (gfx1250) — matrix operand order repaired and checked by reading, compiled, not run on hardware
//
#include <hip/hip_runtime.h>
#include <hip/hip_bf16.h>

// CDNA5 / gfx1250, wave32. WMMA bf16 16x16x32, f32 accumulate.

typedef __attribute__((ext_vector_type(16))) __bf16 v16bf;
typedef __attribute__((ext_vector_type(8)))  float  v8f;

#define D 128
#define LDSK 272   // padded LDS row stride (bf16 elems): 272*2=544B = 17*32B -> banks spread

// ---------------------------------------------------------------------------
// Kernel 1: zero the aggregation scratch (agg[N*128] ++ deg[N], contiguous f32)
// ---------------------------------------------------------------------------
__global__ void k_zero(float* __restrict__ p, long long total) {
    long long i = (long long)blockIdx.x * blockDim.x + threadIdx.x;
    if (i < total) p[i] = 0.0f;
}

// ---------------------------------------------------------------------------
// Kernel 2: edge scatter-add.  One wave32 per edge: lane l handles features
// 4l..4l+3 (float4 gather of the 512B source row, 4 f32 global atomics).
// Atomics land in L2 (192MB holds agg+x easily).
// ---------------------------------------------------------------------------
__global__ void k_scatter(const float* __restrict__ x, const int* __restrict__ ei,
                          float* __restrict__ agg, float* __restrict__ deg,
                          int E, int n) {
    long long t = (long long)blockIdx.x * blockDim.x + threadIdx.x;
    long long e = t >> 5;
    if (e >= E) return;
    int lane = (int)(t & 31);
    int src = ei[e];
    int dst = ei[(long long)E + e];
    src = min(max(src, 0), n - 1);
    dst = min(max(dst, 0), n - 1);
    const float4* xr = (const float4*)(x + (size_t)src * D);
    float4 v = xr[lane];
    float* ad = agg + (size_t)dst * D + lane * 4;
    atomicAdd(ad + 0, v.x);
    atomicAdd(ad + 1, v.y);
    atomicAdd(ad + 2, v.z);
    atomicAdd(ad + 3, v.w);
    if (lane == 0) atomicAdd(deg + dst, 1.0f);
}

// ---------------------------------------------------------------------------
// Kernel 3: fused  conv = [mean_nb | x] @ [W_l ; W_r]^T + b_l,
//           out = LayerNorm(relu(conv + x)) * gamma + beta
// 256 threads = 8 waves; wave w owns 16 node rows. K = 256 (8 slabs of 32).
// B (combined weights, bf16) staged once per block in LDS.
// ---------------------------------------------------------------------------
__global__ __launch_bounds__(256) void k_gnn_wmma(
    const float* __restrict__ x,   const float* __restrict__ Wl,
    const float* __restrict__ bl,  const float* __restrict__ Wr,
    const float* __restrict__ gamma, const float* __restrict__ beta,
    const float* __restrict__ agg, const float* __restrict__ deg,
    float* __restrict__ out, int n)
{
    extern __shared__ __bf16 sW[];   // [128][LDSK] : sW[nf][k], k<128 = W_l, k>=128 = W_r

    // cooperative stage of combined weight matrix, f32 -> bf16
    for (int idx = threadIdx.x; idx < 128 * 256; idx += 256) {
        int nf = idx >> 8;
        int k  = idx & 255;
        float w = (k < 128) ? Wl[nf * 128 + k] : Wr[nf * 128 + (k - 128)];
        sW[nf * LDSK + k] = (__bf16)w;
    }
    __syncthreads();

    const int wave = threadIdx.x >> 5;
    const int lane = threadIdx.x & 31;
    const int half = lane >> 4;      // K-half / row-half selector
    const int ml   = lane & 15;      // row (for A) / column (for B,C,D)

    const int rowBase = blockIdx.x * 128 + wave * 16;
    const int gr  = rowBase + ml;            // this lane's A row
    const int grc = min(gr, n - 1);          // clamp: keep EXEC all-ones for WMMA

    const float dg   = deg[grc];
    const float invd = 1.0f / fmaxf(dg, 1.0f);

    // -------- build A fragments (kept live across all 8 N-tiles) ----------
    // lane layout: element e -> K = kc + half*16 + e
    v16bf afrag[8];
    const float* aggRow = agg + (size_t)grc * D;
    const float* xRow   = x   + (size_t)grc * D;
    #pragma unroll
    for (int c = 0; c < 8; ++c) {
        const float* sp = (c < 4) ? aggRow : xRow;
        const float  sc = (c < 4) ? invd : 1.0f;
        const int    kb = (c & 3) * 32 + half * 16;
        v16bf a;
        #pragma unroll
        for (int e = 0; e < 8; ++e) {
            a[e]     = (__bf16)(sp[kb + e]      * sc);
            a[e + 8] = (__bf16)(sp[kb + 8 + e] * sc);
        }
        afrag[c] = a;
    }

    // -------- 8 N-tiles x 8 K-slabs of v_wmma_f32_16x16x32_bf16 -----------
    v8f acc[8];
    #pragma unroll
    for (int t = 0; t < 8; ++t) {
        v8f c = {};
        // B fragment base: column = t*16 + ml, K-half = half, 16 contiguous K
        const __bf16* bp = sW + (size_t)(t * 16 + ml) * LDSK + half * 16;
        #pragma unroll
        for (int kc = 0; kc < 8; ++kc) {
            v16bf b = *(const v16bf*)(bp + kc * 32);
            c = __builtin_amdgcn_wmma_f32_16x16x32_bf16(
                    /*neg_a=*/false, afrag[kc], /*neg_b=*/false, b,
                    /*c_mod=*/(short)0, c, /*reuse_a=*/false, /*reuse_b=*/false);
        }
        acc[t] = c;
    }

    // -------- epilogue: +b_l + residual, relu, LayerNorm, store ------------
    // D layout: acc[t][r] is (row = r + 8*half, col = t*16 + ml)
    float sum[8], sq[8];
    #pragma unroll
    for (int r = 0; r < 8; ++r) { sum[r] = 0.0f; sq[r] = 0.0f; }

    #pragma unroll
    for (int r = 0; r < 8; ++r) {
        const int gm  = rowBase + 8 * half + r;
        const int gmc = min(gm, n - 1);
        const float* xres = x + (size_t)gmc * D;
        #pragma unroll
        for (int t = 0; t < 8; ++t) {
            const int col = t * 16 + ml;
            float v = acc[t][r] + bl[col] + xres[col];
            v = fmaxf(v, 0.0f);
            acc[t][r] = v;
            sum[r] += v;
            sq[r]  += v * v;
        }
    }

    #pragma unroll
    for (int r = 0; r < 8; ++r) {
        float s = sum[r], q = sq[r];
        // row lives entirely in one 16-lane half: xor masks 1..8 stay in-group
        #pragma unroll
        for (int m = 1; m < 16; m <<= 1) {
            s += __shfl_xor(s, m, 32);
            q += __shfl_xor(q, m, 32);
        }
        const float mu  = s * (1.0f / 128.0f);
        const float var = q * (1.0f / 128.0f) - mu * mu;
        const float rs  = rsqrtf(var + 1e-5f);
        const int gm = rowBase + 8 * half + r;
        if (gm < n) {
            float* orow = out + (size_t)gm * D;
            #pragma unroll
            for (int t = 0; t < 8; ++t) {
                const int col = t * 16 + ml;
                orow[col] = (acc[t][r] - mu) * rs * gamma[col] + beta[col];
            }
        }
    }
}

// ---------------------------------------------------------------------------
extern "C" void kernel_launch(void* const* d_in, const int* in_sizes, int n_in,
                              void* d_out, int out_size, void* d_ws, size_t ws_size,
                              hipStream_t stream) {
    const float* x     = (const float*)d_in[0];
    const int*   ei    = (const int*)  d_in[1];
    const float* Wl    = (const float*)d_in[2];
    const float* bl    = (const float*)d_in[3];
    const float* Wr    = (const float*)d_in[4];
    const float* gamma = (const float*)d_in[5];
    const float* beta  = (const float*)d_in[6];
    float*       out   = (float*)d_out;

    const int n = in_sizes[0] / D;     // 100,000
    const int E = in_sizes[1] / 2;     // 1,000,000

    float* agg = (float*)d_ws;                 // [n*128] f32
    float* deg = agg + (size_t)n * D;          // [n]     f32

    // 1) zero scratch
    const long long ztot = (long long)n * D + n;
    {
        long long blocks = (ztot + 255) / 256;
        k_zero<<<dim3((unsigned)blocks), 256, 0, stream>>>(agg, ztot);
    }

    // 2) edge scatter-add (one wave32 per edge)
    {
        long long threads = (long long)E * 32;
        long long blocks  = (threads + 255) / 256;
        k_scatter<<<dim3((unsigned)blocks), 256, 0, stream>>>(x, ei, agg, deg, E, n);
    }

    // 3) fused WMMA GEMM + epilogue (128 rows / block, 8 waves)
    {
        int blocks = (n + 127) / 128;
        size_t shmem = (size_t)128 * LDSK * sizeof(__bf16);   // 69,632 B
        k_gnn_wmma<<<dim3(blocks), 256, shmem, stream>>>(
            x, Wl, bl, Wr, gamma, beta, agg, deg, out, n);
    }
}
